// GRUODEClassifier_31765578121839
// MI455X (gfx1250) — matrix operand order repaired and checked by reading, not compiled, not run on hardware
//
#include <hip/hip_runtime.h>
#include <hip/hip_bf16.h>
#include <math.h>

// ---------------------------------------------------------------------------
// GRU-ODE classifier for MI455X (gfx1250), wave32 + WMMA bf16.
// B=256, L=1024, F=64, E=H=128, 3H=384, ODE_STEPS=4.
// Grid: 16 blocks (one per 16-row batch tile), 256 threads = 8 waves.
// Wave w owns output columns [16w,16w+16) of every 128-wide activation and
// gate sub-block. Weights live in LDS in pre-swizzled B-fragment layout.
// LN/head reduction is intra-wave (shfl_xor within 16-lane groups): no
// barriers and no LDS round-trips on that stage.
// ---------------------------------------------------------------------------

typedef __attribute__((ext_vector_type(16))) __bf16 v16bf;
typedef __attribute__((ext_vector_type(8)))  __bf16 v8bf;
typedef __attribute__((ext_vector_type(8)))  float  v8f;

#define B_   256
#define L_   1024
#define F_   64
#define H_   128
#define G_   384
#define NSTEP 4

// ---- workspace layout (halves for packed bf16 weights) --------------------
#define OWX   0                    // 64x128  -> 8192 halves
#define OW1   8192                 // 128x128 -> 16384
#define OW2   24576                // 128x128 -> 16384
#define OWIH  40960                // 128x384 -> 49152
#define OWHH  90112                // 128x384 -> 49152
#define W_HALVES 139264            // total packed halves (278528 bytes)
#define OANY_BYTES (W_HALVES * 2)  // int any_active[L] after weights

// ---- dynamic LDS layout (bytes) -------------------------------------------
constexpr int OFF_X    = 278528;             // bf16 [16][72]
constexpr int OFF_XE   = OFF_X   + 2304;     // bf16 [16][136]
constexpr int OFF_TMP  = OFF_XE  + 4352;     // bf16 [16][136]
constexpr int OFF_H    = OFF_TMP + 4352;     // bf16 [16][136]
constexpr int OFF_HF   = OFF_H   + 4352;     // f32  [16][132]
constexpr int OFF_BX   = OFF_HF  + 8448;     // f32 [128]
constexpr int OFF_B1   = OFF_BX  + 512;
constexpr int OFF_B2   = OFF_B1  + 512;
constexpr int OFF_BIH  = OFF_B2  + 512;      // f32 [384]
constexpr int OFF_BHH  = OFF_BIH + 1536;
constexpr int OFF_LNG  = OFF_BHH + 1536;
constexpr int OFF_LNB  = OFF_LNG + 512;
constexpr int OFF_WHD  = OFF_LNB + 512;
constexpr int OFF_DTM  = OFF_WHD + 512;      // f32 [16]
constexpr int SMEM_TOTAL = OFF_DTM + 64;     // ~308,544 B  (< 320 KB/WGP)
static_assert(SMEM_TOTAL <= 320 * 1024, "exceeds WGP LDS");

// ---------------------------------------------------------------------------
static __device__ __forceinline__ v8f vzero() {
    v8f z;
#pragma unroll
    for (int i = 0; i < 8; ++i) z[i] = 0.f;
    return z;
}

static __device__ __forceinline__ v16bf combine16(v8bf lo, v8bf hi) {
    v16bf r;
#pragma unroll
    for (int i = 0; i < 8; ++i) { r[i] = lo[i]; r[i + 8] = hi[i]; }
    return r;
}

// A fragment (16x32 bf16) from row-major padded LDS tile [16][stride].
// lane<16: row=lane,  K = kb+{0..15}
// lane>=16: row=lane-16, K = kb+{16..31}
static __device__ __forceinline__ v16bf loadA(const __bf16* base, int stride, int kb) {
    const int lane = threadIdx.x & 31;
    const __bf16* p = base + (lane & 15) * stride + kb + ((lane >> 4) << 4);
    v8bf lo = *(const v8bf*)p;
    v8bf hi = *(const v8bf*)(p + 8);
    return combine16(lo, hi);
}

// B fragment (32x16 bf16) from packed fragment-major LDS blob:
// fragment f occupies 32 lanes * 16 halves, contiguous.
static __device__ __forceinline__ v16bf loadB(const __bf16* packed, int frag) {
    const int lane = threadIdx.x & 31;
    const __bf16* p = packed + (((frag << 5) + lane) << 4);
    v8bf lo = *(const v8bf*)p;
    v8bf hi = *(const v8bf*)(p + 8);
    return combine16(lo, hi);
}

static __device__ __forceinline__ v8f wmma_bf16(v16bf a, v16bf b, v8f c) {
    return __builtin_amdgcn_wmma_f32_16x16x32_bf16(false, a, false, b,
                                                   (short)0, c, false, false);
}

// D tile (16x16 f32) scatter: lane<16 -> rows 0..7, lane>=16 -> rows 8..15
static __device__ __forceinline__ void storeTileBF16(__bf16* base, int stride,
                                                     int ncol0, const v8f& c) {
    const int lane = threadIdx.x & 31;
    const int m0 = (lane >> 4) << 3;
    const int n = ncol0 + (lane & 15);
#pragma unroll
    for (int r = 0; r < 8; ++r) base[(m0 + r) * stride + n] = (__bf16)c[r];
}

static __device__ __forceinline__ void storeTileF32(float* base, int stride,
                                                    int ncol0, const v8f& c) {
    const int lane = threadIdx.x & 31;
    const int m0 = (lane >> 4) << 3;
    const int n = ncol0 + (lane & 15);
#pragma unroll
    for (int r = 0; r < 8; ++r) base[(m0 + r) * stride + n] = c[r];
}

static __device__ __forceinline__ float sigm(float v) {
    return 1.f / (1.f + __expf(-v));
}

// ---------------------------------------------------------------------------
// Prep: pack weight W[N][K] (row-major f32) into bf16 B-fragment layout.
// dst[((nt*KF + kf)*32 + lane)*16 + i] = W[n][k],
//   n = nt*16 + (lane&15), k = kf*32 + (lane>>4)*16 + i.
// ---------------------------------------------------------------------------
__global__ void pack_b_kernel(const float* __restrict__ W, __bf16* __restrict__ dst,
                              int K, int N) {
    int id = blockIdx.x * blockDim.x + threadIdx.x;
    if (id >= K * N) return;
    int i    = id & 15;
    int lane = (id >> 4) & 31;
    int fidx = id >> 9;
    int KF = K >> 5;
    int kf = fidx % KF;
    int nt = fidx / KF;
    int n = nt * 16 + (lane & 15);
    int k = kf * 32 + ((lane >> 4) << 4) + i;
    dst[id] = (__bf16)W[n * K + k];
}

// Prep: any_active[t] = OR over full batch of mask[b][t] (reference semantics).
__global__ void any_mask_kernel(const int* __restrict__ mask, int* __restrict__ any) {
    int t = blockIdx.x * blockDim.x + threadIdx.x;
    if (t >= L_) return;
    int a = 0;
    for (int b = 0; b < B_; ++b) a |= mask[b * L_ + t];
    any[t] = (a != 0);
}

// ---------------------------------------------------------------------------
// Main persistent scan kernel.
// ---------------------------------------------------------------------------
__global__ void __launch_bounds__(256) gruode_main(
    const float* __restrict__ x,   const float* __restrict__ dt,
    const int*   __restrict__ mask,
    const float* __restrict__ bx,  const float* __restrict__ b1,
    const float* __restrict__ b2,  const float* __restrict__ bih,
    const float* __restrict__ bhh, const float* __restrict__ lng,
    const float* __restrict__ lnb, const float* __restrict__ whd,
    const float* __restrict__ bhp, const __bf16* __restrict__ wpack,
    const int*   __restrict__ anyact, float* __restrict__ out) {
    extern __shared__ __align__(16) char smem[];

    const int tid  = threadIdx.x;
    const int lane = tid & 31;
    const int w    = tid >> 5;          // wave id 0..7  -> N-tile base 16*w
    const int b0   = blockIdx.x * 16;   // batch tile base
    const int row  = tid >> 4;          // 0..15: row handled in LN stage
    const int seg  = tid & 15;          // 0..15: 8-col segment in LN stage

    __bf16* sWgt = (__bf16*)smem;
    __bf16* sWX  = sWgt + OWX;
    __bf16* sW1  = sWgt + OW1;
    __bf16* sW2  = sWgt + OW2;
    __bf16* sWIH = sWgt + OWIH;
    __bf16* sWHH = sWgt + OWHH;
    __bf16* sX   = (__bf16*)(smem + OFF_X);
    __bf16* sXE  = (__bf16*)(smem + OFF_XE);
    __bf16* sTMP = (__bf16*)(smem + OFF_TMP);
    __bf16* sH   = (__bf16*)(smem + OFF_H);
    float*  sHF  = (float*)(smem + OFF_HF);
    float*  sBX  = (float*)(smem + OFF_BX);
    float*  sB1  = (float*)(smem + OFF_B1);
    float*  sB2  = (float*)(smem + OFF_B2);
    float*  sBIH = (float*)(smem + OFF_BIH);
    float*  sBHH = (float*)(smem + OFF_BHH);
    float*  sLNG = (float*)(smem + OFF_LNG);
    float*  sLNB = (float*)(smem + OFF_LNB);
    float*  sWHd = (float*)(smem + OFF_WHD);
    float*  sDTM = (float*)(smem + OFF_DTM);

    // One-time: copy packed weights (272KB) into LDS; load biases; zero h.
    {
        const uint4* src = (const uint4*)wpack;
        uint4* dst = (uint4*)smem;
        for (int i = tid; i < W_HALVES / 8; i += 256) dst[i] = src[i];
        for (int i = tid; i < 128; i += 256) {
            sBX[i] = bx[i]; sB1[i] = b1[i]; sB2[i] = b2[i];
            sLNG[i] = lng[i]; sLNB[i] = lnb[i]; sWHd[i] = whd[i];
        }
        for (int i = tid; i < 384; i += 256) { sBIH[i] = bih[i]; sBHH[i] = bhh[i]; }
        for (int i = tid; i < 16 * 136; i += 256) sH[i] = (__bf16)0.f;
    }
    __syncthreads();

    const float bhv = bhp[0];
    const int   colL = lane & 15;
    const int   m0   = (lane >> 4) << 3;

    v8f hacc = vzero();            // this wave's 16x16 f32 tile of h
    float accv = 0.f, cntv = 0.f;  // per-row logit accumulators (16 per row)

#pragma unroll 1
    for (int t = 0; t < L_; ++t) {
        // ---- stage 1: stream x[:, t, :], dt, mask into LDS --------------
        float mf;  // this thread's row mask (register; avoids LDS WAR hazard)
        {
            int m = tid >> 4, f4 = (tid & 15) * 4;
            const float* xp = x + ((size_t)(b0 + m) * L_ + t) * F_ + f4;
            const float4 xv = *(const float4*)xp;
            // prefetch next timestep's x tile into cache (global_prefetch_b8)
            if (t + 1 < L_) __builtin_prefetch(xp + F_, 0, 3);
            __bf16* dp = sX + m * 72 + f4;
            dp[0] = (__bf16)xv.x; dp[1] = (__bf16)xv.y;
            dp[2] = (__bf16)xv.z; dp[3] = (__bf16)xv.w;
            mf = (float)mask[(size_t)(b0 + row) * L_ + t];
            if (tid < 16) {
                float d   = dt[(size_t)(b0 + tid) * L_ + t];
                float mfl = (float)mask[(size_t)(b0 + tid) * L_ + t];
                sDTM[tid] = d * mfl * (1.f / NSTEP);
            }
        }
        const int act = anyact[t];
        __syncthreads();

        const v8f hprev = hacc;

        // ---- xe = relu(x @ Wx^T + bx) -----------------------------------
        {
            v8f c = vzero();
#pragma unroll
            for (int kf = 0; kf < 2; ++kf)
                c = wmma_bf16(loadA(sX, 72, kf * 32), loadB(sWX, w * 2 + kf), c);
            float bv = sBX[w * 16 + colL];
#pragma unroll
            for (int r = 0; r < 8; ++r) c[r] = fmaxf(c[r] + bv, 0.f);
            storeTileBF16(sXE, 136, w * 16, c);   // synced by ODE barriers
        }

        // ---- ODE Euler: 4x  h += step * (tanh(h W1^T + b1) W2^T + b2) ---
        float dts[8];
#pragma unroll
        for (int r = 0; r < 8; ++r) dts[r] = sDTM[m0 + r];
        const float b1v = sB1[w * 16 + colL];
        const float b2v = sB2[w * 16 + colL];
#pragma unroll 1
        for (int s = 0; s < NSTEP; ++s) {
            v8f c1 = vzero();
#pragma unroll
            for (int kf = 0; kf < 4; ++kf)
                c1 = wmma_bf16(loadA(sH, 136, kf * 32), loadB(sW1, w * 4 + kf), c1);
#pragma unroll
            for (int r = 0; r < 8; ++r) c1[r] = tanhf(c1[r] + b1v);
            storeTileBF16(sTMP, 136, w * 16, c1);
            __syncthreads();
            v8f c2 = vzero();
#pragma unroll
            for (int kf = 0; kf < 4; ++kf)
                c2 = wmma_bf16(loadA(sTMP, 136, kf * 32), loadB(sW2, w * 4 + kf), c2);
#pragma unroll
            for (int r = 0; r < 8; ++r) hacc[r] += dts[r] * (c2[r] + b2v);
            storeTileBF16(sH, 136, w * 16, hacc);
            __syncthreads();
        }

        // ---- GRU cell (wave w handles cols 16w of r/z/n blocks) ---------
        {
            v8f gir = vzero(), giz = vzero(), gin = vzero();
            v8f ghr = vzero(), ghz = vzero(), ghn = vzero();
#pragma unroll
            for (int kf = 0; kf < 4; ++kf) {
                v16bf axe = loadA(sXE, 136, kf * 32);
                gir = wmma_bf16(axe, loadB(sWIH, (0 + w) * 4 + kf), gir);
                giz = wmma_bf16(axe, loadB(sWIH, (8 + w) * 4 + kf), giz);
                gin = wmma_bf16(axe, loadB(sWIH, (16 + w) * 4 + kf), gin);
                v16bf ah = loadA(sH, 136, kf * 32);
                ghr = wmma_bf16(ah, loadB(sWHH, (0 + w) * 4 + kf), ghr);
                ghz = wmma_bf16(ah, loadB(sWHH, (8 + w) * 4 + kf), ghz);
                ghn = wmma_bf16(ah, loadB(sWHH, (16 + w) * 4 + kf), ghn);
            }
            const int col = w * 16 + colL;
            const float bir = sBIH[col], biz = sBIH[128 + col], bin = sBIH[256 + col];
            const float bhr = sBHH[col], bhz = sBHH[128 + col], bhn = sBHH[256 + col];
#pragma unroll
            for (int r = 0; r < 8; ++r) {
                float rg = sigm(gir[r] + bir + ghr[r] + bhr);
                float zg = sigm(giz[r] + biz + ghz[r] + bhz);
                float ng = tanhf(gin[r] + bin + rg * (ghn[r] + bhn));
                float hN = (1.f - zg) * ng + zg * hacc[r];
                hacc[r] = act ? hN : hprev[r];
            }
            storeTileBF16(sH, 136, w * 16, hacc);
            storeTileF32(sHF, 132, w * 16, hacc);
        }
        __syncthreads();

        // ---- fused LayerNorm + head, intra-wave (16-lane) reductions ----
        // Row `row`'s 16 segments live in one 16-lane group of one wave, so
        // the reduction over segments is barrier-free via shfl_xor.
        {
            const float* hr = sHF + row * 132 + seg * 8;
            float s = 0.f, s2 = 0.f;
#pragma unroll
            for (int c = 0; c < 8; ++c) { float v = hr[c]; s += v; s2 += v * v; }
#pragma unroll
            for (int d = 1; d < 16; d <<= 1) {
                s  += __shfl_xor(s,  d, 16);
                s2 += __shfl_xor(s2, d, 16);
            }
            const float mu  = s * (1.f / 128.f);
            const float var = s2 * (1.f / 128.f) - mu * mu;
            const float rs  = rsqrtf(var + 1e-5f);
            float p = 0.f;
#pragma unroll
            for (int c = 0; c < 8; ++c) {
                int col = seg * 8 + c;
                float hn = (hr[c] - mu) * rs * sLNG[col] + sLNB[col];
                p += hn * sWHd[col];
            }
#pragma unroll
            for (int d = 1; d < 16; d <<= 1) p += __shfl_xor(p, d, 16);
            accv += (p + bhv) * mf;
            cntv += mf;
        }
        // next iteration's stage-1 barrier protects sX/sDTM reuse
    }

    if (seg == 0) out[b0 + row] = accv / fmaxf(cntv, 1.f);
}

// ---------------------------------------------------------------------------
extern "C" void kernel_launch(void* const* d_in, const int* in_sizes, int n_in,
                              void* d_out, int out_size, void* d_ws, size_t ws_size,
                              hipStream_t stream) {
    (void)in_sizes; (void)n_in; (void)out_size; (void)ws_size;
    const float* x    = (const float*)d_in[0];
    const float* dt   = (const float*)d_in[1];
    const int*   mask = (const int*)d_in[2];
    const float* Wx   = (const float*)d_in[3];
    const float* bx   = (const float*)d_in[4];
    const float* W1   = (const float*)d_in[5];
    const float* b1   = (const float*)d_in[6];
    const float* W2   = (const float*)d_in[7];
    const float* b2   = (const float*)d_in[8];
    const float* Wih  = (const float*)d_in[9];
    const float* bih  = (const float*)d_in[10];
    const float* Whh  = (const float*)d_in[11];
    const float* bhh  = (const float*)d_in[12];
    const float* lng  = (const float*)d_in[13];
    const float* lnb  = (const float*)d_in[14];
    const float* whd  = (const float*)d_in[15];
    const float* bhp  = (const float*)d_in[16];
    float* out = (float*)d_out;

    __bf16* wsH = (__bf16*)d_ws;
    int* anyp = (int*)((char*)d_ws + OANY_BYTES);

    // Pack weights into bf16 B-fragment layout (deterministic every call).
    pack_b_kernel<<<(F_ * H_ + 255) / 256, 256, 0, stream>>>(Wx,  wsH + OWX,  F_, H_);
    pack_b_kernel<<<(H_ * H_ + 255) / 256, 256, 0, stream>>>(W1,  wsH + OW1,  H_, H_);
    pack_b_kernel<<<(H_ * H_ + 255) / 256, 256, 0, stream>>>(W2,  wsH + OW2,  H_, H_);
    pack_b_kernel<<<(H_ * G_ + 255) / 256, 256, 0, stream>>>(Wih, wsH + OWIH, H_, G_);
    pack_b_kernel<<<(H_ * G_ + 255) / 256, 256, 0, stream>>>(Whh, wsH + OWHH, H_, G_);
    any_mask_kernel<<<L_ / 256, 256, 0, stream>>>(mask, anyp);

    hipFuncSetAttribute(reinterpret_cast<const void*>(&gruode_main),
                        hipFuncAttributeMaxDynamicSharedMemorySize, SMEM_TOTAL);
    gruode_main<<<B_ / 16, 256, SMEM_TOTAL, stream>>>(
        x, dt, mask, bx, b1, b2, bih, bhh, lng, lnb, whd, bhp, wsH, anyp, out);
}
